// haar_channel_block_15951508538084
// MI455X (gfx1250) — matrix operand order repaired and checked by reading, not compiled, not run on hardware
//
#include <hip/hip_runtime.h>
#include <hip/hip_bf16.h>
#include <math.h>

// ---------------------------------------------------------------------------
// Types for CDNA5 WMMA
// ---------------------------------------------------------------------------
typedef __attribute__((ext_vector_type(16))) __bf16 bf16x16;
typedef __attribute__((ext_vector_type(8)))  float  f32x8;

union FragB16 { uint4 q[2]; bf16x16 v; };

__device__ __forceinline__ void split_f32(float x, __bf16& hi, __bf16& lo) {
    hi = (__bf16)x;
    lo = (__bf16)(x - (float)hi);
}

#define RSQRT2F 0.70710678118654752440f

// ---------------------------------------------------------------------------
// Kernel 1: transpose + split weights to bf16 hi/lo  (W is KxN, WT is NxK)
// ---------------------------------------------------------------------------
__global__ __launch_bounds__(256) void wprep_kernel(
    const float* __restrict__ W, __bf16* __restrict__ Thi,
    __bf16* __restrict__ Tlo, int K, int N)
{
    int idx = blockIdx.x * 256 + threadIdx.x;
    if (idx >= K * N) return;
    int n = idx / K;
    int k = idx - n * K;
    float v = W[(size_t)k * N + n];
    __bf16 hi, lo;
    split_f32(v, hi, lo);
    Thi[idx] = hi;
    Tlo[idx] = lo;
}

// ---------------------------------------------------------------------------
// Kernel 2: Haar DWT along L for 8 channels per block, output bf16 hi/lo rows
//   H[row = b*512 + c][l], DWT ordering: concat([haar(h), g]) per level.
// ---------------------------------------------------------------------------
__global__ __launch_bounds__(256) void haar_kernel(
    const float* __restrict__ x, __bf16* __restrict__ Hhi, __bf16* __restrict__ Hlo)
{
    __shared__ float bufA[8][514];
    __shared__ float bufB[8][514];
    const int c0  = blockIdx.x * 8;
    const int b   = blockIdx.y;
    const int tid = threadIdx.x;

    // load tile: bufA[c][l] = x[b, l, c0+c]
    for (int idx = tid; idx < 8 * 512; idx += 256) {
        int l = idx >> 3, c = idx & 7;
        bufA[c][l] = x[((size_t)b * 512 + l) * 512 + c0 + c];
    }
    __syncthreads();

    float (*src)[514] = bufA;
    float (*dst)[514] = bufB;
    for (int lh = 8; lh >= 0; --lh) {          // segment length n = 2<<lh
        const int half  = 1 << lh;
        const int total = 8 * half;
        for (int idx = tid; idx < total; idx += 256) {
            int c = idx >> lh;
            int i = idx & (half - 1);
            float e = src[c][2 * i];
            float o = src[c][2 * i + 1];
            float h = (e + o) * RSQRT2F;
            float g = (e - o) * RSQRT2F;
            dst[c][i] = h;
            size_t ro = ((size_t)b * 512 + c0 + c) * 512;
            __bf16 ghi, glo;
            split_f32(g, ghi, glo);
            Hhi[ro + half + i] = ghi;          // g is final at [n/2, n)
            Hlo[ro + half + i] = glo;
            if (lh == 0) {                     // last level: h final at 0
                __bf16 hhi, hlo;
                split_f32(h, hhi, hlo);
                Hhi[ro] = hhi;
                Hlo[ro] = hlo;
            }
        }
        __syncthreads();
        float (*t)[514] = src; src = dst; dst = t;
    }
}

// ---------------------------------------------------------------------------
// Kernel 3/4: split-precision bf16x3 GEMM  C(MxN) = A(MxK) * W(KxN)
//   A given as bf16 hi/lo (row-major), W given transposed (NxK) bf16 hi/lo.
//   EPI==0 : ReLU, re-split to bf16 hi/lo outputs (feeds GEMM2)
//   EPI==1 : sigmoid, f32 output
//   Block: 256 threads (8 waves), tile 128(M) x 64(N), K-step 32.
// ---------------------------------------------------------------------------
template <int EPI>
__global__ __launch_bounds__(256) void gemm_bf16x3_kernel(
    const __bf16* __restrict__ Ahi, const __bf16* __restrict__ Alo,
    const __bf16* __restrict__ Bhi, const __bf16* __restrict__ Blo,
    int K, int N,
    __bf16* __restrict__ OutHi, __bf16* __restrict__ OutLo,
    float* __restrict__ OutF)
{
    __shared__ __align__(16) __bf16 lAhi[128][40];   // stride 80B (16B aligned)
    __shared__ __align__(16) __bf16 lAlo[128][40];
    __shared__ __align__(16) __bf16 lBhi[64][40];
    __shared__ __align__(16) __bf16 lBlo[64][40];

    const int tid   = threadIdx.x;
    const int lane  = tid & 31;
    const int wave  = tid >> 5;
    const int wm    = wave & 3;       // 4 waves along M (32 rows each)
    const int wn    = wave >> 2;      // 2 waves along N (32 cols each)
    const int lrow  = lane & 15;
    const int khalf = lane >> 4;      // 0: lanes 0-15, 1: lanes 16-31
    const int m0    = blockIdx.y * 128;
    const int n0    = blockIdx.x * 64;

    f32x8 acc[2][2] = {};

    for (int k0 = 0; k0 < K; k0 += 32) {
        __syncthreads();
        // stage A tile: 128 rows x 32 halves (hi+lo), 16B vector fills
#pragma unroll
        for (int rep = 0; rep < 2; ++rep) {
            int chunk = rep * 256 + tid;
            int r  = chunk >> 2;
            int cc = (chunk & 3) * 8;
            size_t g = (size_t)(m0 + r) * K + k0 + cc;
            *(uint4*)&lAhi[r][cc] = *(const uint4*)(Ahi + g);
            *(uint4*)&lAlo[r][cc] = *(const uint4*)(Alo + g);
        }
        // stage B tile: 64 cols x 32 halves (hi+lo)
        {
            int col = tid >> 2;
            int cc  = (tid & 3) * 8;
            size_t g = (size_t)(n0 + col) * K + k0 + cc;
            *(uint4*)&lBhi[col][cc] = *(const uint4*)(Bhi + g);
            *(uint4*)&lBlo[col][cc] = *(const uint4*)(Blo + g);
        }
        __syncthreads();

        // fragments per ISA VGPR layouts (16-bit A 16x32, B 32x16)
        FragB16 ahi[2], alo[2], bhi[2], blo[2];
#pragma unroll
        for (int i = 0; i < 2; ++i) {
            int row = wm * 32 + i * 16 + lrow;
            int kb  = khalf * 8;
            ahi[i].q[0] = *(const uint4*)&lAhi[row][kb];
            ahi[i].q[1] = *(const uint4*)&lAhi[row][kb + 16];
            alo[i].q[0] = *(const uint4*)&lAlo[row][kb];
            alo[i].q[1] = *(const uint4*)&lAlo[row][kb + 16];
            int col = wn * 32 + i * 16 + lrow;
            int ko  = khalf * 8;
            bhi[i].q[0] = *(const uint4*)&lBhi[col][ko];
            bhi[i].q[1] = *(const uint4*)&lBhi[col][ko + 16];
            blo[i].q[0] = *(const uint4*)&lBlo[col][ko];
            blo[i].q[1] = *(const uint4*)&lBlo[col][ko + 16];
        }
#pragma unroll
        for (int i = 0; i < 2; ++i)
#pragma unroll
            for (int j = 0; j < 2; ++j) {
                // a*b ~= lo*hi + hi*lo + hi*hi, accumulated in f32
                acc[i][j] = __builtin_amdgcn_wmma_f32_16x16x32_bf16(
                    false, alo[i].v, false, bhi[j].v, (short)0, acc[i][j], false, false);
                acc[i][j] = __builtin_amdgcn_wmma_f32_16x16x32_bf16(
                    false, ahi[i].v, false, blo[j].v, (short)0, acc[i][j], false, false);
                acc[i][j] = __builtin_amdgcn_wmma_f32_16x16x32_bf16(
                    false, ahi[i].v, false, bhi[j].v, (short)0, acc[i][j], false, false);
            }
    }

    // epilogue: C/D layout — VGPR v: M = v (+8 for lanes 16-31), N = lane%16
#pragma unroll
    for (int i = 0; i < 2; ++i)
#pragma unroll
        for (int j = 0; j < 2; ++j)
#pragma unroll
            for (int v = 0; v < 8; ++v) {
                int m = m0 + wm * 32 + i * 16 + khalf * 8 + v;
                int n = n0 + wn * 32 + j * 16 + lrow;
                float val = acc[i][j][v];
                size_t o = (size_t)m * N + n;
                if (EPI == 0) {
                    float r = fmaxf(val, 0.0f);
                    __bf16 hi, lo;
                    split_f32(r, hi, lo);
                    OutHi[o] = hi;
                    OutLo[o] = lo;
                } else {
                    OutF[o] = 1.0f / (1.0f + __expf(-val));
                }
            }
}

// ---------------------------------------------------------------------------
// Kernel 5: per-row mean / rstd of S (rows of 512), wave-per-row
// ---------------------------------------------------------------------------
__global__ __launch_bounds__(256) void rowstats_kernel(
    const float* __restrict__ S, float* __restrict__ mu, float* __restrict__ rstd)
{
    int wave = threadIdx.x >> 5;
    int lane = threadIdx.x & 31;
    int row  = blockIdx.x * 8 + wave;
    const float* p = S + (size_t)row * 512;
    float s = 0.0f, ss = 0.0f;
#pragma unroll
    for (int i = 0; i < 16; ++i) {
        float v = p[lane + i * 32];
        s  += v;
        ss += v * v;
    }
#pragma unroll
    for (int off = 16; off >= 1; off >>= 1) {
        s  += __shfl_xor(s,  off, 32);
        ss += __shfl_xor(ss, off, 32);
    }
    if (lane == 0) {
        float m   = s * (1.0f / 512.0f);
        float var = ss * (1.0f / 512.0f) - m * m;
        mu[row]   = m;
        rstd[row] = rsqrtf(var + 1e-6f);
    }
}

// ---------------------------------------------------------------------------
// Kernel 6: LN (gamma/beta indexed by l!) + transpose + multiply by x
//   out[b,l,c] = x[b,l,c] * ((S[(b,c),l]-mu)*rstd*gamma[l] + beta[l])
// ---------------------------------------------------------------------------
__global__ __launch_bounds__(256) void finalize_kernel(
    const float* __restrict__ S, const float* __restrict__ x,
    const float* __restrict__ gamma, const float* __restrict__ beta,
    const float* __restrict__ mu, const float* __restrict__ rstd,
    float* __restrict__ out)
{
    __shared__ float t[64][65];
    const int l0 = blockIdx.x * 64;
    const int c0 = blockIdx.y * 64;
    const int b  = blockIdx.z;
    const int tid = threadIdx.x;

#pragma unroll
    for (int it = 0; it < 16; ++it) {
        int idx = it * 256 + tid;
        int c = idx >> 6, l = idx & 63;
        int row = b * 512 + c0 + c;
        float sv = S[(size_t)row * 512 + l0 + l];
        t[c][l] = (sv - mu[row]) * rstd[row] * gamma[l0 + l] + beta[l0 + l];
    }
    __syncthreads();
#pragma unroll
    for (int it = 0; it < 16; ++it) {
        int idx = it * 256 + tid;
        int l = idx >> 6, c = idx & 63;
        size_t o = ((size_t)b * 512 + l0 + l) * 512 + c0 + c;
        out[o] = x[o] * t[c][l];
    }
}

// ---------------------------------------------------------------------------
// Host launcher
// ---------------------------------------------------------------------------
extern "C" void kernel_launch(void* const* d_in, const int* in_sizes, int n_in,
                              void* d_out, int out_size, void* d_ws, size_t ws_size,
                              hipStream_t stream)
{
    const float* x     = (const float*)d_in[0];
    const float* w1    = (const float*)d_in[1];
    const float* w2    = (const float*)d_in[2];
    const float* gamma = (const float*)d_in[3];
    const float* beta  = (const float*)d_in[4];
    float* out = (float*)d_out;

    char* ws = (char*)d_ws;
    const size_t MB = 1ull << 20;
    // H (hi/lo) lives in [0, 64MB); S reuses [0, 64MB) after H is dead.
    __bf16* Hhi   = (__bf16*)(ws + 0 * MB);      // 32 MB
    __bf16* Hlo   = (__bf16*)(ws + 32 * MB);     // 32 MB
    float*  S     = (float*) (ws + 0 * MB);      // 64 MB (overlaps dead H)
    __bf16* W1hi  = (__bf16*)(ws + 64 * MB);     // 1 MB each
    __bf16* W1lo  = (__bf16*)(ws + 65 * MB);
    __bf16* W2hi  = (__bf16*)(ws + 66 * MB);
    __bf16* W2lo  = (__bf16*)(ws + 67 * MB);
    __bf16* Hmhi  = (__bf16*)(ws + 68 * MB);     // 64 MB
    __bf16* Hmlo  = (__bf16*)(ws + 132 * MB);    // 64 MB
    float*  mu    = (float*) (ws + 196 * MB);    // 128 KB
    float*  rstd  = (float*) (ws + 196 * MB + 128 * 1024);

    const int M = 64 * 512;   // 32768 rows

    // 1) weight prep (transpose + bf16 split)
    wprep_kernel<<<(512 * 1024 + 255) / 256, 256, 0, stream>>>(w1, W1hi, W1lo, 512, 1024);
    wprep_kernel<<<(1024 * 512 + 255) / 256, 256, 0, stream>>>(w2, W2hi, W2lo, 1024, 512);

    // 2) Haar DWT of x^T rows
    haar_kernel<<<dim3(512 / 8, 64), 256, 0, stream>>>(x, Hhi, Hlo);

    // 3) GEMM1 + ReLU : (32768x512) @ (512x1024)
    gemm_bf16x3_kernel<0><<<dim3(1024 / 64, M / 128), 256, 0, stream>>>(
        Hhi, Hlo, W1hi, W1lo, 512, 1024, Hmhi, Hmlo, nullptr);

    // 4) GEMM2 + sigmoid : (32768x1024) @ (1024x512)
    gemm_bf16x3_kernel<1><<<dim3(512 / 64, M / 128), 256, 0, stream>>>(
        Hmhi, Hmlo, W2hi, W2lo, 1024, 512, nullptr, nullptr, S);

    // 5) row statistics for LN
    rowstats_kernel<<<M / 8, 256, 0, stream>>>(S, mu, rstd);

    // 6) LN + transpose + multiply with x
    finalize_kernel<<<dim3(8, 8, 64), 256, 0, stream>>>(S, x, gamma, beta, mu, rstd, out);
}
